// ColorHistograms_49976239456336
// MI455X (gfx1250) — matrix operand order repaired and checked by reading, not compiled, not run on hardware
//
#include <hip/hip_runtime.h>
#include <hip/hip_bf16.h>

typedef __attribute__((ext_vector_type(16))) _Float16 v16h;
typedef __attribute__((ext_vector_type(8)))  _Float16 v8h;
typedef __attribute__((ext_vector_type(8)))  float    v8f;

#define NBT   8192          // B*T frames
#define HBINS 512
#define TT    1024          // T
#define KPAD  128           // padded window width (101 -> 128)

// Assemble a 16-half WMMA fragment from two contiguous 8-half (16B) chunks.
__device__ __forceinline__ v16h make_frag(const _Float16* p0, const _Float16* p1) {
    v8h lo = *(const v8h*)p0;
    v8h hi = *(const v8h*)p1;
    v16h f;
#pragma unroll
    for (int i = 0; i < 8; ++i) { f[i] = lo[i]; f[i + 8] = hi[i]; }
    return f;
}

// ---------------------------------------------------------------------------
// Kernel 0a: zero the padded windows buffer (8192 x 128 f16 = 2 MB).
__global__ void zero_win_kernel(uint4* __restrict__ p) {
    uint4 z; z.x = 0u; z.y = 0u; z.z = 0u; z.w = 0u;
    p[(size_t)blockIdx.x * 256 + threadIdx.x] = z;
}

// Kernel 0b: convert W_fc [128,101] f32 -> Wh [128,128] f16, zero-padded.
__global__ void conv_w_kernel(const float* __restrict__ Wfc, _Float16* __restrict__ Wh) {
    int idx = blockIdx.x * 256 + threadIdx.x;      // 0 .. 16383
    int o = idx >> 7, k = idx & 127;
    float v = (k < 101) ? Wfc[o * 101 + k] : 0.0f;
    Wh[idx] = (_Float16)v;
}

// ---------------------------------------------------------------------------
// Kernel 1: per-frame 512-bin histogram + L2 normalize -> f16 x[8192,512].
// One block per frame, 256 threads, LDS atomics.
__global__ void hist_kernel(const int* __restrict__ frames, _Float16* __restrict__ xh) {
    __shared__ unsigned int hist[HBINS];
    __shared__ float red[256];
    const int f   = blockIdx.x;
    const int tid = threadIdx.x;

    hist[tid]       = 0u;
    hist[tid + 256] = 0u;
    __syncthreads();

    const int* fp = frames + (size_t)f * 3072;   // 1024 px * 3 ch
#pragma unroll
    for (int p = 0; p < 4; ++p) {
        int pix = p * 256 + tid;
        int r = fp[3 * pix + 0];
        int g = fp[3 * pix + 1];
        int b = fp[3 * pix + 2];
        int bin = ((r >> 5) << 6) | ((g >> 5) << 3) | (b >> 5);
        atomicAdd(&hist[bin], 1u);
    }
    __syncthreads();

    float c0 = (float)hist[tid];
    float c1 = (float)hist[tid + 256];
    red[tid] = c0 * c0 + c1 * c1;
    __syncthreads();
    for (int s = 128; s > 0; s >>= 1) {
        if (tid < s) red[tid] += red[tid + s];
        __syncthreads();
    }
    float scale = 1.0f / fmaxf(sqrtf(red[0]), 1e-12f);

    _Float16* out = xh + (size_t)f * HBINS;
    out[tid]       = (_Float16)(c0 * scale);
    out[tid + 256] = (_Float16)(c1 * scale);
}

// ---------------------------------------------------------------------------
// Kernel 2: banded Gram matrix via WMMA. One wave per (batch, row-tile I).
// All 9 J-tiles are computed unconditionally (out-of-range tiles clamped to a
// valid tile -> harmless work, never scattered), so there is NO control flow
// around the WMMAs and EXEC stays all-1s with pure scalar loop structure.
__global__ void band_kernel(const _Float16* __restrict__ xh, _Float16* __restrict__ win) {
    const int lane = threadIdx.x & 31;
    // Wave index is uniform: pin it to an SGPR so all derived control/addresses
    // that don't involve `lane` stay scalar.
    const int gwi  = __builtin_amdgcn_readfirstlane(blockIdx.x * 8 + (threadIdx.x >> 5));
    const int b    = gwi >> 6;                // batch 0..7
    const int I    = gwi & 63;                // row tile 0..63
    const int half = lane >> 4;
    const int l16  = lane & 15;

    const _Float16* xb = xh + (size_t)b * TT * HBINS;

    v8f acc[9];
#pragma unroll
    for (int j = 0; j < 9; ++j) acc[j] = (v8f){};

    // A fragment base: lane l16 supplies row 16*I + l16 (per ISA 16-bit A layout).
    const _Float16* aptr = xb + (size_t)(I * 16 + l16) * HBINS + half * 8;

    // Per-tile B row bases (clamped J), hoisted out of the k loop.
    const _Float16* bptr[9];
#pragma unroll
    for (int j = 0; j < 9; ++j) {
        int J  = I - 4 + j;
        int Jc = J < 0 ? 0 : (J > 63 ? 63 : J);   // scalar clamp, no branches
        bptr[j] = xb + (size_t)(Jc * 16 + l16) * HBINS + half * 8;
    }

    for (int kc = 0; kc < 16; ++kc) {
        const int kb = kc * 32;
        v16h afrag = make_frag(aptr + kb, aptr + kb + 16);
#pragma unroll
        for (int j = 0; j < 9; ++j) {
            v16h bfrag = make_frag(bptr[j] + kb, bptr[j] + kb + 16);
            acc[j] = __builtin_amdgcn_wmma_f32_16x16x32_f16(
                false, afrag, false, bfrag, (short)0, acc[j], false, false);
        }
    }

    // Scatter band to windows (D layout: n = lane&15, m = r + 8*(lane>>4)).
    const int mBase = I * 16;
#pragma unroll
    for (int j = 0; j < 9; ++j) {
        int J = I - 4 + j;
        if (J < 0 || J >= 64) continue;           // scalar branch (I is SGPR)
        int s = J * 16 + l16;
#pragma unroll
        for (int r = 0; r < 8; ++r) {
            int m = mBase + r + half * 8;
            int w = s - m + 50;
            if (w >= 0 && w <= 100) {
                win[((size_t)b * TT + m) * KPAD + w] = (_Float16)acc[j][r];
            }
        }
    }
}

// ---------------------------------------------------------------------------
// Kernel 3: FC via WMMA: out[8192,128] = relu(windows[8192,128] @ Wh^T + b).
// One wave per (16-row tile Im, 16-col tile Jn) -> 512*8 = 4096 waves.
__global__ void fc_kernel(const _Float16* __restrict__ win, const _Float16* __restrict__ Wh,
                          const float* __restrict__ bfc, float* __restrict__ out) {
    const int lane = threadIdx.x & 31;
    const int gwi  = __builtin_amdgcn_readfirstlane(blockIdx.x * 8 + (threadIdx.x >> 5));
    const int Im   = gwi >> 3;                // 0 .. 511
    const int Jn   = gwi & 7;                 // 0 .. 7
    const int half = lane >> 4;
    const int l16  = lane & 15;

    v8f acc = (v8f){};
    const _Float16* aptr = win + (size_t)(Im * 16 + l16) * KPAD + half * 8;
    const _Float16* bptr = Wh  + (size_t)(Jn * 16 + l16) * KPAD + half * 8;

#pragma unroll
    for (int kc = 0; kc < 4; ++kc) {
        const int kb = kc * 32;
        v16h afrag = make_frag(aptr + kb, aptr + kb + 16);
        v16h bfrag = make_frag(bptr + kb, bptr + kb + 16);
        acc = __builtin_amdgcn_wmma_f32_16x16x32_f16(
            false, afrag, false, bfrag, (short)0, acc, false, false);
    }

    const int n = Jn * 16 + l16;
    const float bias = bfc[n];
#pragma unroll
    for (int r = 0; r < 8; ++r) {
        int m = Im * 16 + r + half * 8;
        float v = acc[r] + bias;
        out[(size_t)m * 128 + n] = v > 0.0f ? v : 0.0f;
    }
}

// ---------------------------------------------------------------------------
extern "C" void kernel_launch(void* const* d_in, const int* in_sizes, int n_in,
                              void* d_out, int out_size, void* d_ws, size_t ws_size,
                              hipStream_t stream) {
    (void)in_sizes; (void)n_in; (void)out_size; (void)ws_size;
    const int*   frames = (const int*)d_in[0];    // [8,1024,32,32,3] int32
    const float* Wfc    = (const float*)d_in[1];  // [128,101] f32
    const float* bfc    = (const float*)d_in[2];  // [128] f32
    float*       out    = (float*)d_out;          // [8,1024,128] f32

    char* ws = (char*)d_ws;
    _Float16* xh  = (_Float16*)ws;                               // 8192*512*2  = 8 MB
    _Float16* win = (_Float16*)(ws + (size_t)NBT * HBINS * 2);   // 8192*128*2  = 2 MB
    _Float16* Wh  = (_Float16*)(ws + (size_t)NBT * HBINS * 2
                                   + (size_t)NBT * KPAD * 2);    // 128*128*2   = 32 KB

    zero_win_kernel<<<dim3((NBT * KPAD * 2) / 16 / 256), dim3(256), 0, stream>>>((uint4*)win);
    conv_w_kernel<<<dim3(64), dim3(256), 0, stream>>>(Wfc, Wh);
    hist_kernel<<<dim3(NBT), dim3(256), 0, stream>>>(frames, xh);
    band_kernel<<<dim3(64), dim3(256), 0, stream>>>(xh, win);
    fc_kernel<<<dim3(512), dim3(256), 0, stream>>>(win, Wh, bfc, out);
}
